// MultiheadAttention_146028888610
// MI455X (gfx1250) — matrix operand order repaired and checked by reading, not compiled, not run on hardware
//
#include <hip/hip_runtime.h>
#include <hip/hip_bf16.h>
#include <cstdint>
#include <cstddef>

// ---------------------------------------------------------------------------
// MI455X (gfx1250) fused multi-head attention forward.
// B=2, S=2048, W=1024, H=16, d=64.  All GEMMs via v_wmma_f32_16x16x32_f16.
// Double-buffered LDS staging: one barrier per K-slab, global loads for the
// next slab overlap the WMMAs of the current one.
// ---------------------------------------------------------------------------

typedef __attribute__((ext_vector_type(16))) _Float16 v16h;
typedef __attribute__((ext_vector_type(8)))  _Float16 v8h;
typedef __attribute__((ext_vector_type(8)))  float    v8f;

static constexpr int Bb = 2, Ss = 2048, Ww = 1024, Hh = 16, Dd = 64;
static constexpr float ATTN_SCALE = 0.35355339059327378f;  // 64^(-1/4)

__device__ __forceinline__ v8f wmma_f16(v16h a, v16h b, v8f c) {
  // D = A(16x32 f16) * B(32x16 f16) + C(16x16 f32)
  return __builtin_amdgcn_wmma_f32_16x16x32_f16(false, a, false, b, (short)0, c,
                                                false, false);
}

// A-fragment (16x32, f16) from a row-major tile with leading dim `ld` halfs.
// Lane l holds row (l&15); VGPRs 0-3 = K {0..7 | 8..15}, VGPRs 4-7 = +16.
__device__ __forceinline__ v16h load_afrag(const _Float16* p, int ld, int lane) {
  const _Float16* q = p + (lane & 15) * ld + ((lane >> 4) << 3);
  v8h lo = *(const v8h*)(q);
  v8h hi = *(const v8h*)(q + 16);
  v16h out;
#pragma unroll
  for (int i = 0; i < 8; ++i) { out[i] = lo[i]; out[i + 8] = hi[i]; }
  return out;
}

// B-fragment (32x16, f16) from B^T stored row-major [N][K] with leading dim `ld`.
// Lane l holds column (l&15); VGPRs 0-3 = K {0..7 | 8..15}, VGPRs 4-7 = +16.
__device__ __forceinline__ v16h load_bfrag(const _Float16* p, int ld, int lane) {
  const _Float16* q = p + (lane & 15) * ld + ((lane >> 4) << 3);
  v8h lo = *(const v8h*)(q);
  v8h hi = *(const v8h*)(q + 16);
  v16h out;
#pragma unroll
  for (int i = 0; i < 8; ++i) { out[i] = lo[i]; out[i + 8] = hi[i]; }
  return out;
}

// ------------------------------- utility kernels ---------------------------

__global__ __launch_bounds__(256) void cvt_f2h(const float* __restrict__ in,
                                               _Float16* __restrict__ out, int n) {
  int i = blockIdx.x * 256 + threadIdx.x;
  if (i < n) out[i] = (_Float16)in[i];
}

// in: f32 [K][N] -> out: f16 [N][K]
__global__ void transpose_cvt(const float* __restrict__ in,
                              _Float16* __restrict__ out, int K, int N) {
  __shared__ float tile[32][33];
  const int k0 = blockIdx.y * 32, n0 = blockIdx.x * 32;
  const int tx = threadIdx.x, ty = threadIdx.y;
  for (int i = ty; i < 32; i += 8)
    tile[i][tx] = in[(size_t)(k0 + i) * N + n0 + tx];
  __syncthreads();
  for (int i = ty; i < 32; i += 8)
    out[(size_t)(n0 + i) * K + k0 + tx] = (_Float16)tile[tx][i];
}

// per-z slice: in f16 [R][C] -> out f16 [C][R]
__global__ void transpose_h(const _Float16* __restrict__ in,
                            _Float16* __restrict__ out, int R, int C) {
  __shared__ _Float16 tile[32][33];
  const int r0 = blockIdx.y * 32, c0 = blockIdx.x * 32;
  const size_t base = (size_t)blockIdx.z * R * C;
  const int tx = threadIdx.x, ty = threadIdx.y;
  for (int i = ty; i < 32; i += 8)
    tile[i][tx] = in[base + (size_t)(r0 + i) * C + c0 + tx];
  __syncthreads();
  for (int i = ty; i < 32; i += 8)
    out[base + (size_t)(c0 + i) * R + r0 + tx] = tile[tx][i];
}

// ------------------------------- GEMM mainloop -----------------------------
// C(128x128) tile per block, 8 waves in 4x2, each wave 32x64 (2x4 WMMA tiles).
// As/Bs are double buffers: 2 * 128*40 halfs each.
__device__ __forceinline__ void gemm_mainloop(const _Float16* __restrict__ A,
                                              const _Float16* __restrict__ Bt,
                                              int K, int m0, int n0,
                                              _Float16* As, _Float16* Bs,
                                              v8f acc[2][4]) {
  const int tid = threadIdx.x, lane = tid & 31, wave = tid >> 5;
  const int wm = wave >> 1, wn = wave & 1;
  const int lr = tid >> 1, lk = (tid & 1) * 16;

#pragma unroll
  for (int i = 0; i < 2; ++i)
#pragma unroll
    for (int j = 0; j < 4; ++j)
#pragma unroll
      for (int e = 0; e < 8; ++e) acc[i][j][e] = 0.0f;

  const _Float16* ga = A + (size_t)(m0 + lr) * K + lk;
  const _Float16* gb = Bt + (size_t)(n0 + lr) * K + lk;

  {  // preload slab 0 into buffer 0
    const uint4* pa = (const uint4*)(ga);
    const uint4* pb = (const uint4*)(gb);
    uint4* la = (uint4*)(As + lr * 40 + lk);
    uint4* lb = (uint4*)(Bs + lr * 40 + lk);
    la[0] = pa[0]; la[1] = pa[1];
    lb[0] = pb[0]; lb[1] = pb[1];
  }
  __syncthreads();

  for (int kb = 0; kb < K; kb += 32) {
    const int cur = (kb >> 5) & 1;
    const _Float16* Ac = As + cur * (128 * 40);
    const _Float16* Bc = Bs + cur * (128 * 40);
    const bool hasNext = (kb + 32) < K;

    uint4 na0, na1, nb0, nb1;  // next slab fetched during compute
    if (hasNext) {
      const uint4* pa = (const uint4*)(ga + kb + 32);
      const uint4* pb = (const uint4*)(gb + kb + 32);
      na0 = pa[0]; na1 = pa[1];
      nb0 = pb[0]; nb1 = pb[1];
      if (kb + 64 < K) {  // gfx1250 global_prefetch_b8 two slabs ahead
        __builtin_prefetch(ga + kb + 64, 0, 1);
        __builtin_prefetch(gb + kb + 64, 0, 1);
      }
    }

    v16h af[2], bf[4];
#pragma unroll
    for (int i = 0; i < 2; ++i)
      af[i] = load_afrag(Ac + (wm * 32 + i * 16) * 40, 40, lane);
#pragma unroll
    for (int j = 0; j < 4; ++j)
      bf[j] = load_bfrag(Bc + (wn * 64 + j * 16) * 40, 40, lane);
#pragma unroll
    for (int i = 0; i < 2; ++i)
#pragma unroll
      for (int j = 0; j < 4; ++j) acc[i][j] = wmma_f16(af[i], bf[j], acc[i][j]);

    if (hasNext) {
      _Float16* An = As + (cur ^ 1) * (128 * 40);
      _Float16* Bn = Bs + (cur ^ 1) * (128 * 40);
      uint4* la = (uint4*)(An + lr * 40 + lk);
      uint4* lb = (uint4*)(Bn + lr * 40 + lk);
      la[0] = na0; la[1] = na1;
      lb[0] = nb0; lb[1] = nb1;
    }
    __syncthreads();
  }
}

// ----------------------- QKV projection + head scatter ---------------------

__global__ __launch_bounds__(256) void gemm_qkv(
    const _Float16* __restrict__ A,    // [B*S][W]
    const _Float16* __restrict__ Bt,   // [3W][W]
    const float* __restrict__ bias,    // [3W]
    _Float16* __restrict__ qh, _Float16* __restrict__ kh,
    _Float16* __restrict__ vtmp) {
  __shared__ __align__(16) _Float16 As[2 * 128 * 40];
  __shared__ __align__(16) _Float16 Bs[2 * 128 * 40];
  const int lane = threadIdx.x & 31, wave = threadIdx.x >> 5;
  const int wm = wave >> 1, wn = wave & 1;
  const int m0 = blockIdx.y * 128, n0 = blockIdx.x * 128;

  v8f acc[2][4];
  gemm_mainloop(A, Bt, Ww, m0, n0, As, Bs, acc);

  const int hf = lane >> 4, ln = lane & 15;
#pragma unroll
  for (int i = 0; i < 2; ++i) {
#pragma unroll
    for (int j = 0; j < 4; ++j) {
      const int gn0 = n0 + wn * 64 + j * 16;     // column within [0,3W)
      const int h = gn0 / 192;                   // head (192 = 3*64 per head)
      const int part = (gn0 % 192) / 64;         // 0=q 1=k 2=v
      const int c = (gn0 & 63) + ln;             // channel within head
      const float bval = bias[gn0 + ln];
#pragma unroll
      for (int r = 0; r < 8; ++r) {
        const int m = m0 + wm * 32 + i * 16 + r + 8 * hf;
        const int bidx = m >> 11;                // m / S
        const int s = m & (Ss - 1);
        const float v = acc[i][j][r] + bval;
        const size_t dst = ((size_t)(bidx * Hh + h) * Ss + s) * Dd + c;
        if (part == 0)      qh[dst] = (_Float16)(v * ATTN_SCALE);
        else if (part == 1) kh[dst] = (_Float16)(v * ATTN_SCALE);
        else                vtmp[dst] = (_Float16)v;
      }
    }
  }
}

// ------------------------- flash-style attention ---------------------------
// Block: one (b,h), 128 queries; 8 waves * 16 queries. Key blocks of 64,
// K/V tiles double-buffered in LDS (one barrier per key block).

__global__ __launch_bounds__(256) void attn_kernel(
    const _Float16* __restrict__ qh,   // [BH][S][64], pre-scaled
    const _Float16* __restrict__ kh,   // [BH][S][64], pre-scaled
    const _Float16* __restrict__ vt,   // [BH][64][S]
    _Float16* __restrict__ attnO) {    // [B][S][H*64]
  __shared__ __align__(16) _Float16 Qs[128 * 72];
  __shared__ __align__(16) _Float16 Ks[2 * 64 * 72];
  __shared__ __align__(16) _Float16 Vs[2 * 64 * 72];
  __shared__ __align__(16) _Float16 Ps[8][16 * 72];

  const int bh = blockIdx.y;
  const int q0 = blockIdx.x * 128;
  const int tid = threadIdx.x, lane = tid & 31, wave = tid >> 5;
  const _Float16* qg = qh + (size_t)bh * Ss * Dd;
  const _Float16* kg = kh + (size_t)bh * Ss * Dd;
  const _Float16* vg = vt + (size_t)bh * Ss * Dd;

  const int rr = tid >> 2, sg4 = (tid & 3) * 16;  // K/V staging pattern

  {  // stage Q tile [128][64]
    const int r = tid >> 1, sg = (tid & 1) * 32;
    const uint4* g = (const uint4*)(qg + (size_t)(q0 + r) * Dd + sg);
    uint4* l = (uint4*)(Qs + r * 72 + sg);
    l[0] = g[0]; l[1] = g[1]; l[2] = g[2]; l[3] = g[3];
  }
  {  // preload first K tile [64][64] and V^T tile [64][64] into buffer 0
    const uint4* gk = (const uint4*)(kg + (size_t)rr * Dd + sg4);
    uint4* lk = (uint4*)(Ks + rr * 72 + sg4);
    lk[0] = gk[0]; lk[1] = gk[1];
    const uint4* gv = (const uint4*)(vg + (size_t)rr * Ss + sg4);
    uint4* lv = (uint4*)(Vs + rr * 72 + sg4);
    lv[0] = gv[0]; lv[1] = gv[1];
  }
  __syncthreads();

  float mrow[8], lrow[8];
  v8f oacc[4];
#pragma unroll
  for (int r = 0; r < 8; ++r) { mrow[r] = -3.0e38f; lrow[r] = 0.0f; }
#pragma unroll
  for (int j = 0; j < 4; ++j)
#pragma unroll
    for (int e = 0; e < 8; ++e) oacc[j][e] = 0.0f;

  const int hf = lane >> 4, ln = lane & 15;
  _Float16* ps = &Ps[wave][0];

  for (int kb = 0; kb < Ss; kb += 64) {
    const int cur = (kb >> 6) & 1;
    const _Float16* Kc = Ks + cur * (64 * 72);
    const _Float16* Vc = Vs + cur * (64 * 72);
    const bool hasNext = (kb + 64) < Ss;

    uint4 nk0, nk1, nv0, nv1;  // next K/V tile fetched during compute
    if (hasNext) {
      const uint4* gk = (const uint4*)(kg + (size_t)(kb + 64 + rr) * Dd + sg4);
      nk0 = gk[0]; nk1 = gk[1];
      const uint4* gv = (const uint4*)(vg + (size_t)rr * Ss + kb + 64 + sg4);
      nv0 = gv[0]; nv1 = gv[1];
      if (kb + 128 < Ss) {
        __builtin_prefetch(kg + (size_t)(kb + 128 + rr) * Dd + sg4, 0, 1);
        __builtin_prefetch(vg + (size_t)rr * Ss + kb + 128 + sg4, 0, 1);
      }
    }

    // logits = Q(16x64) * K^T(64x64): 4 n-tiles x 2 k-steps
    v16h aq0 = load_afrag(Qs + (wave * 16) * 72 + 0, 72, lane);
    v16h aq1 = load_afrag(Qs + (wave * 16) * 72 + 32, 72, lane);
    v8f lg[4];
#pragma unroll
    for (int j = 0; j < 4; ++j) {
#pragma unroll
      for (int e = 0; e < 8; ++e) lg[j][e] = 0.0f;
      v16h b0 = load_bfrag(Kc + (j * 16) * 72 + 0, 72, lane);
      v16h b1 = load_bfrag(Kc + (j * 16) * 72 + 32, 72, lane);
      lg[j] = wmma_f16(aq0, b0, lg[j]);
      lg[j] = wmma_f16(aq1, b1, lg[j]);
    }

    // online softmax (rows live in half-waves: reduce over 16 lanes)
#pragma unroll
    for (int r = 0; r < 8; ++r) {
      float mx = fmaxf(fmaxf(lg[0][r], lg[1][r]), fmaxf(lg[2][r], lg[3][r]));
#pragma unroll
      for (int d = 1; d < 16; d <<= 1) mx = fmaxf(mx, __shfl_xor(mx, d, 32));
      const float mnew = fmaxf(mrow[r], mx);
      const float alpha = __expf(mrow[r] - mnew);
      mrow[r] = mnew;
      float psum = 0.0f;
#pragma unroll
      for (int j = 0; j < 4; ++j) {
        const float p = __expf(lg[j][r] - mnew);
        lg[j][r] = p;
        psum += p;
      }
      lrow[r] = lrow[r] * alpha + psum;  // per-lane partial row sum
#pragma unroll
      for (int j = 0; j < 4; ++j) oacc[j][r] *= alpha;
    }

    // C-layout -> A-fragment layout via per-wave LDS round trip
#pragma unroll
    for (int j = 0; j < 4; ++j)
#pragma unroll
      for (int r = 0; r < 8; ++r)
        ps[(r + 8 * hf) * 72 + j * 16 + ln] = (_Float16)lg[j][r];

    v16h ap0 = load_afrag(ps + 0, 72, lane);
    v16h ap1 = load_afrag(ps + 32, 72, lane);
#pragma unroll
    for (int j = 0; j < 4; ++j) {  // out += P(16x64) * V(64x64)
      v16h b0 = load_bfrag(Vc + (j * 16) * 72 + 0, 72, lane);
      v16h b1 = load_bfrag(Vc + (j * 16) * 72 + 32, 72, lane);
      oacc[j] = wmma_f16(ap0, b0, oacc[j]);
      oacc[j] = wmma_f16(ap1, b1, oacc[j]);
    }

    if (hasNext) {  // write next K/V tile into the other buffer
      _Float16* Kn = Ks + (cur ^ 1) * (64 * 72);
      _Float16* Vn = Vs + (cur ^ 1) * (64 * 72);
      uint4* lk = (uint4*)(Kn + rr * 72 + sg4);
      lk[0] = nk0; lk[1] = nk1;
      uint4* lv = (uint4*)(Vn + rr * 72 + sg4);
      lv[0] = nv0; lv[1] = nv1;
    }
    __syncthreads();
  }

  const int bidx = bh >> 4, h = bh & 15;
#pragma unroll
  for (int r = 0; r < 8; ++r) {
    float sum = lrow[r];
#pragma unroll
    for (int d = 1; d < 16; d <<= 1) sum += __shfl_xor(sum, d, 32);
    const float inv = 1.0f / sum;
    const int sq = q0 + wave * 16 + r + 8 * hf;
    const size_t row = (size_t)(bidx * Ss + sq) * Ww + h * Dd;
#pragma unroll
    for (int j = 0; j < 4; ++j)
      attnO[row + j * 16 + ln] = (_Float16)(oacc[j][r] * inv);
  }
}

// ------------------------------ output projection --------------------------

__global__ __launch_bounds__(256) void gemm_proj(
    const _Float16* __restrict__ A,   // [B*S][W]
    const _Float16* __restrict__ Bt,  // [W][W]
    const float* __restrict__ bias, float* __restrict__ out) {
  __shared__ __align__(16) _Float16 As[2 * 128 * 40];
  __shared__ __align__(16) _Float16 Bs[2 * 128 * 40];
  const int lane = threadIdx.x & 31, wave = threadIdx.x >> 5;
  const int wm = wave >> 1, wn = wave & 1;
  const int m0 = blockIdx.y * 128, n0 = blockIdx.x * 128;

  v8f acc[2][4];
  gemm_mainloop(A, Bt, Ww, m0, n0, As, Bs, acc);

  const int hf = lane >> 4, ln = lane & 15;
#pragma unroll
  for (int i = 0; i < 2; ++i) {
#pragma unroll
    for (int j = 0; j < 4; ++j) {
      const int gn0 = n0 + wn * 64 + j * 16;
      const float bval = bias[gn0 + ln];
#pragma unroll
      for (int r = 0; r < 8; ++r) {
        const int m = m0 + wm * 32 + i * 16 + r + 8 * hf;
        out[(size_t)m * Ww + gn0 + ln] = acc[i][j][r] + bval;
      }
    }
  }
}

// ------------------------------- launcher ----------------------------------

extern "C" void kernel_launch(void* const* d_in, const int* in_sizes, int n_in,
                              void* d_out, int out_size, void* d_ws,
                              size_t ws_size, hipStream_t stream) {
  (void)in_sizes; (void)n_in; (void)out_size; (void)ws_size;
  const float* x      = (const float*)d_in[0];
  const float* W_qkv  = (const float*)d_in[1];
  const float* b_qkv  = (const float*)d_in[2];
  const float* W_proj = (const float*)d_in[3];
  const float* b_proj = (const float*)d_in[4];
  float* out = (float*)d_out;

  // f16 workspace layout (halfs), with liveness-based reuse (total 40 MiB):
  _Float16* ws     = (_Float16*)d_ws;
  _Float16* xh     = ws;              // 4M  x(f16);        reused as vt later
  _Float16* wqkvT  = ws + 4194304;    // 3M  W_qkv^T
  _Float16* wprojT = ws + 7340032;    // 1M  W_proj^T
  _Float16* qh     = ws + 8388608;    // 4M  q [BH][S][64] (scaled)
  _Float16* kh     = ws + 12582912;   // 4M  k [BH][S][64] (scaled)
  _Float16* vtmp   = ws + 16777216;   // 4M  v [BH][S][64]; reused as attn out
  _Float16* vt     = xh;              //     v^T [BH][64][S]
  _Float16* attn   = vtmp;            //     attention output [B][S][H*64]

  const int M = Bb * Ss;  // 4096

  cvt_f2h<<<dim3((M * Ww) / 256), dim3(256), 0, stream>>>(x, xh, M * Ww);
  transpose_cvt<<<dim3(3 * Ww / 32, Ww / 32), dim3(32, 8), 0, stream>>>(
      W_qkv, wqkvT, Ww, 3 * Ww);
  transpose_cvt<<<dim3(Ww / 32, Ww / 32), dim3(32, 8), 0, stream>>>(
      W_proj, wprojT, Ww, Ww);
  gemm_qkv<<<dim3(3 * Ww / 128, M / 128), dim3(256), 0, stream>>>(
      xh, wqkvT, b_qkv, qh, kh, vtmp);
  transpose_h<<<dim3(Dd / 32, Ss / 32, Bb * Hh), dim3(32, 8), 0, stream>>>(
      vtmp, vt, Ss, Dd);
  attn_kernel<<<dim3(Ss / 128, Bb * Hh), dim3(256), 0, stream>>>(
      qh, kh, vt, attn);
  gemm_proj<<<dim3(Ww / 128, M / 128), dim3(256), 0, stream>>>(
      attn, wprojT, b_proj, out);
}
